// BitLinear_51900384805156
// MI455X (gfx1250) — matrix operand order repaired and checked by reading, not compiled, not run on hardware
//
#include <hip/hip_runtime.h>
#include <hip/hip_bf16.h>

// ---------------------------------------------------------------------------
// BitLinear (8-bit fake-quant linear):  y = x @ qw^T + qb
//   qw = round(clamp(w,-1,1)*255)/255  -> store n = round(...) exactly in f16
//   GEMM: v_wmma_f32_16x16x32_f16, f32 accumulate, scale by 1/255 in epilogue
// M=16384, K=4096, N=4096 -> compute bound (AI ~ 950 flop/byte)
// Double-buffered LDS pipeline, async global->LDS for weights, 1 barrier/iter.
// ---------------------------------------------------------------------------

typedef __attribute__((ext_vector_type(16))) _Float16 v16h;
typedef __attribute__((ext_vector_type(8)))  _Float16 v8h;
typedef __attribute__((ext_vector_type(4)))  _Float16 v4h;
typedef __attribute__((ext_vector_type(8)))  float    v8f;
typedef __attribute__((ext_vector_type(4)))  int      v4i;

typedef __attribute__((address_space(1))) v4i gv4i;  // global int4
typedef __attribute__((address_space(3))) v4i lv4i;  // LDS int4

#if defined(__has_builtin)
#if __has_builtin(__builtin_amdgcn_global_load_async_to_lds_b128)
#define USE_ASYNC_LDS 1
#endif
#endif
#ifndef USE_ASYNC_LDS
#define USE_ASYNC_LDS 0
#endif

union AFrag { v16h v; v8h h[2]; };

static __device__ __forceinline__ void wait_async0() {
#if USE_ASYNC_LDS
#if defined(__has_builtin) && __has_builtin(__builtin_amdgcn_s_wait_asynccnt)
  __builtin_amdgcn_s_wait_asynccnt(0);
#else
  asm volatile("s_wait_asynccnt 0" ::: "memory");
#endif
#endif
}

// ---------------------------------------------------------------------------
// Pass 1: quantize weights:  w16[o*K+i] = (f16) round(clamp(w,-1,1)*255)
// Integer-valued (|n|<=255) -> exact in fp16. Final 1/255 applied in epilogue.
// ---------------------------------------------------------------------------
__global__ __launch_bounds__(256) void quant_weight_kernel(
    const float4* __restrict__ w, v4h* __restrict__ w16, int n4) {
  int i = blockIdx.x * blockDim.x + threadIdx.x;
  if (i < n4) {
    float4 v = w[i];
    v4h h;
    h[0] = (_Float16)rintf(fminf(fmaxf(v.x, -1.f), 1.f) * 255.f);
    h[1] = (_Float16)rintf(fminf(fmaxf(v.y, -1.f), 1.f) * 255.f);
    h[2] = (_Float16)rintf(fminf(fmaxf(v.z, -1.f), 1.f) * 255.f);
    h[3] = (_Float16)rintf(fminf(fmaxf(v.w, -1.f), 1.f) * 255.f);
    w16[i] = h;
  }
}

// ---------------------------------------------------------------------------
// Pass 2: tiled WMMA GEMM, double-buffered.
//   Block: 256 threads = 8 wave32. Tile 128x128, K-tile 64.
//   Wave grid 2(M) x 4(N): each wave owns 64x32 = 8 accumulators (16x16).
//   LDS rows padded to 72 halves (144B) -> conflict-free b128 fragment loads.
// ---------------------------------------------------------------------------
__global__ __launch_bounds__(256) void bitlinear_gemm_kernel(
    const float* __restrict__ x, const _Float16* __restrict__ w16,
    const float* __restrict__ bias, float* __restrict__ out,
    int M, int K, int N, int mblocks, int nblocks) {
  __shared__ _Float16 As[2][128][72];  // ping-pong: 128(M) x 64(K), pad 8
  __shared__ _Float16 Bs[2][128][72];  // ping-pong: 128(N) x 64(K), pad 8

  const int tid  = threadIdx.x;
  const int lane = tid & 31;
  const int wave = tid >> 5;

  // --- L2-friendly block rasterization: 8 M-blocks per N sweep ---
  const int GM = 8;
  int pid   = blockIdx.x;
  int group = GM * nblocks;
  int g     = pid / group;
  int rem   = pid % group;
  int gmcnt = mblocks - g * GM; if (gmcnt > GM) gmcnt = GM;
  int m0 = (g * GM + rem % gmcnt) * 128;
  int n0 = (rem / gmcnt) * 128;

  const int wrow = (wave & 1) * 64;  // wave M offset in tile
  const int wcol = (wave >> 1) * 32; // wave N offset in tile

  const int lhalf   = lane >> 4;     // 0: lanes 0-15, 1: lanes 16-31
  const int l16     = lane & 15;
  const int abase_k = lhalf * 8;     // A-frag K sub-offset (ISA 7.12.2)
  const int bbase_k = lhalf * 8;     // B-frag K sub-offset

  // loader mappings
  const int ac4 = tid & 15;          // 16 float4 per A row
  const int ar0 = tid >> 4;          // 16 rows per pass (x8)
  const int bc8 = tid & 7;           // 8 b128 per B row
  const int br0 = tid >> 3;          // 32 rows per pass (x4)

  float4 areg[8];                    // global->reg staging for next A tile

  auto load_a_global = [&](int kt) {
#pragma unroll
    for (int i = 0; i < 8; ++i)
      areg[i] = *(const float4*)(x + (size_t)(m0 + ar0 + 16 * i) * K + kt + ac4 * 4);
  };
  auto store_a_lds = [&](int buf) {
#pragma unroll
    for (int i = 0; i < 8; ++i) {
      float4 v = areg[i];
      v4h h = { (_Float16)v.x, (_Float16)v.y, (_Float16)v.z, (_Float16)v.w };
      *(v4h*)&As[buf][ar0 + 16 * i][ac4 * 4] = h;
    }
  };
  auto load_b_async = [&](int kt, int buf) {
#pragma unroll
    for (int i = 0; i < 4; ++i) {
      const _Float16* src = w16 + (size_t)(n0 + br0 + 32 * i) * K + kt + bc8 * 8;
      _Float16* dst = &Bs[buf][br0 + 32 * i][bc8 * 8];
#if USE_ASYNC_LDS
      __builtin_amdgcn_global_load_async_to_lds_b128((gv4i*)src, (lv4i*)dst, 0, 0);
#else
      *(v8h*)dst = *(const v8h*)src;
#endif
    }
  };

  v8f acc[4][2];
#pragma unroll
  for (int mt = 0; mt < 4; ++mt)
#pragma unroll
    for (int nt = 0; nt < 2; ++nt) acc[mt][nt] = (v8f)0.0f;

  // ---- prologue: stage tile 0 into buffer 0 ----
  load_b_async(0, 0);
  load_a_global(0);
  store_a_lds(0);

#pragma unroll 1
  for (int kt = 0; kt < K; kt += 64) {
    const int cur = (kt >> 6) & 1;
    const int nxt = cur ^ 1;

    wait_async0();        // tile kt's B copies done (this wave)
    __syncthreads();      // all waves: tile kt visible, buffer nxt free

    const bool has_next = (kt + 64 < K);
    if (has_next) {
      load_b_async(kt + 64, nxt);  // overlaps with compute below
      load_a_global(kt + 64);      // global latency hidden behind WMMAs
    }

    // ---- compute tile kt: 2 k-steps of 32, 8 WMMAs each ----
#pragma unroll
    for (int kk = 0; kk < 64; kk += 32) {
      AFrag a[4], b[2];
#pragma unroll
      for (int mt = 0; mt < 4; ++mt) {
        int m = wrow + mt * 16 + l16;
        a[mt].h[0] = *(const v8h*)&As[cur][m][kk + abase_k];
        a[mt].h[1] = *(const v8h*)&As[cur][m][kk + 16 + abase_k];
      }
#pragma unroll
      for (int nt = 0; nt < 2; ++nt) {
        int n = wcol + nt * 16 + l16;
        b[nt].h[0] = *(const v8h*)&Bs[cur][n][kk + bbase_k];
        b[nt].h[1] = *(const v8h*)&Bs[cur][n][kk + 16 + bbase_k];
      }
#pragma unroll
      for (int mt = 0; mt < 4; ++mt)
#pragma unroll
        for (int nt = 0; nt < 2; ++nt)
          acc[mt][nt] = __builtin_amdgcn_wmma_f32_16x16x32_f16(
              false, a[mt].v, false, b[nt].v, (short)0, acc[mt][nt],
              false, false);
    }

    if (has_next) store_a_lds(nxt);  // cvt + ds_store staged A for next tile
  }

  // ---- epilogue: scale by 1/255, add quantized bias ----
  const float inv255 = 1.0f / 255.0f;
#pragma unroll
  for (int nt = 0; nt < 2; ++nt) {
    int n = n0 + wcol + nt * 16 + l16;
    float qb = rintf(fminf(fmaxf(bias[n], -1.f), 1.f) * 255.f) * inv255;
#pragma unroll
    for (int mt = 0; mt < 4; ++mt) {
      int mbase = m0 + wrow + mt * 16 + lhalf * 8;
#pragma unroll
      for (int r = 0; r < 8; ++r)
        out[(size_t)(mbase + r) * N + n] = acc[mt][nt][r] * inv255 + qb;
    }
  }
}

// ---------------------------------------------------------------------------
extern "C" void kernel_launch(void* const* d_in, const int* in_sizes, int n_in,
                              void* d_out, int out_size, void* d_ws, size_t ws_size,
                              hipStream_t stream) {
  const float* x    = (const float*)d_in[0];
  const float* w    = (const float*)d_in[1];
  const float* bias = (const float*)d_in[2];
  float* out = (float*)d_out;

  const int DOUT = in_sizes[2];                 // bias length
  const int DIN  = in_sizes[1] / DOUT;          // 4096
  const int M    = in_sizes[0] / DIN;           // B*S = 16384
  const int K    = DIN, N = DOUT;

  // scratch: fp16 integer-valued quantized weight, DOUT*DIN*2 bytes (32 MB)
  _Float16* w16 = (_Float16*)d_ws;

  {
    int n4 = (DOUT * DIN) / 4;
    int blocks = (n4 + 255) / 256;
    quant_weight_kernel<<<blocks, 256, 0, stream>>>((const float4*)w,
                                                    (v4h*)w16, n4);
  }
  {
    int mblocks = M / 128, nblocks = N / 128;
    bitlinear_gemm_kernel<<<mblocks * nblocks, 256, 0, stream>>>(
        x, w16, bias, out, M, K, N, mblocks, nblocks);
  }
}
